// MultiHeadLatentAttention_34067680592384
// MI455X (gfx1250) — matrix operand order repaired and checked by reading, not compiled, not run on hardware
//
#include <hip/hip_runtime.h>
#include <hip/hip_bf16.h>

#define S_LEN 2048
#define HID   2048
#define NHEAD 16
#define QLORA 768
#define KVLORA 512
#define KVPAD 640       // KVLORA + D_ROPE padded to multiple of 128
#define D_NOPE 128
#define D_ROPE 64
#define D_V   128
#define D_QK  192   // D_NOPE + D_ROPE

typedef __attribute__((ext_vector_type(16))) __bf16 v16bf;
typedef __attribute__((ext_vector_type(8)))  __bf16 bhalf8;
typedef __attribute__((ext_vector_type(8)))  float  v8f;

static __device__ __forceinline__ v16bf make_frag(const __bf16* lo, const __bf16* hi) {
  bhalf8 a = *(const bhalf8*)lo;
  bhalf8 b = *(const bhalf8*)hi;
  return __builtin_shufflevector(a, b, 0,1,2,3,4,5,6,7,8,9,10,11,12,13,14,15);
}

// Async copy of 16 bytes global -> LDS (CDNA5 async path, ASYNCcnt-tracked)
static __device__ __forceinline__ void async_b128(const __bf16* gptr, __bf16* lptr) {
  unsigned lds = (unsigned)(size_t)lptr;
  asm volatile("global_load_async_to_lds_b128 %0, %1, off"
               :: "v"(lds), "v"(gptr) : "memory");
}
static __device__ __forceinline__ void async_wait0() {
  asm volatile("s_wait_asynccnt 0" ::: "memory");
}

// ---------------------------------------------------------------------------
// fp32 -> bf16 elementwise convert
// ---------------------------------------------------------------------------
__global__ __launch_bounds__(256)
void cvt_f32_bf16(__bf16* __restrict__ dst, const float* __restrict__ src, int n) {
  int i = (blockIdx.x * 256 + threadIdx.x) * 4;
  if (i + 3 < n) {
    float4 v = *(const float4*)(src + i);
    dst[i + 0] = (__bf16)v.x;
    dst[i + 1] = (__bf16)v.y;
    dst[i + 2] = (__bf16)v.z;
    dst[i + 3] = (__bf16)v.w;
  } else {
    for (int k = i; k < n; ++k) dst[k] = (__bf16)src[k];
  }
}

// fp32 [rows][cols] -> bf16 [rows][cols_pad], zero-padding the tail columns
__global__ __launch_bounds__(256)
void cvt_pad_bf16(__bf16* __restrict__ dst, const float* __restrict__ src,
                  int rows, int cols, int cols_pad) {
  int idx = blockIdx.x * 256 + threadIdx.x;
  if (idx >= rows * cols_pad) return;
  int r = idx / cols_pad, c = idx % cols_pad;
  dst[idx] = (c < cols) ? (__bf16)src[(size_t)r * cols + c] : (__bf16)0.0f;
}

// ---------------------------------------------------------------------------
// bf16 GEMM, fp32 accumulate:  C[M,N] = A[M,K] @ B[K,N]
// Block tile 256x128, 8 waves (wave32) as 4(M) x 2(N), wave tile 64x64,
// K step 32 -> 16 WMMAs per iteration per wave.
// Requirements (all call sites satisfy): M %256==0, K %32==0, N %128==0.
// Double-buffered LDS; A staged via global_load_async_to_lds_b128.
// ---------------------------------------------------------------------------
__global__ __launch_bounds__(256)
void gemm_bf16(float* __restrict__ C, const __bf16* __restrict__ A,
               const __bf16* __restrict__ B, int M, int N, int K) {
  __shared__ __align__(16) __bf16 sA[2][256 * 32];   // [stage][row][k]
  __shared__ __align__(16) __bf16 sBt[2][128 * 32];  // [stage][col][k]

  const int tid   = threadIdx.x;
  const int wave  = tid >> 5;
  const int lane  = tid & 31;
  const int lrow  = lane & 15;
  const int lhalf = lane >> 4;
  const int m0 = blockIdx.y * 256;
  const int n0 = blockIdx.x * 128;
  const int wm = (wave & 3) * 64;
  const int wn = (wave >> 2) * 64;

  // staging indices
  const int bkk = tid >> 4;           // 0..15 (k rows bkk, bkk+16)
  const int bnn = (tid & 15) * 8;     // n chunk

  v8f acc[4][4];
#pragma unroll
  for (int a = 0; a < 4; ++a)
#pragma unroll
    for (int b = 0; b < 4; ++b) acc[a][b] = (v8f)0.0f;

  auto stageA = [&](int k0, int st) {
#pragma unroll
    for (int it = 0; it < 4; ++it) {
      int c   = tid + it * 256;       // 0..1023
      int row = c >> 2;               // 0..255
      int kg  = (c & 3) * 8;
      async_b128(A + (size_t)(m0 + row) * K + k0 + kg, &sA[st][row * 32 + kg]);
    }
  };
  auto stageB = [&](int k0, int st) {
#pragma unroll
    for (int it = 0; it < 2; ++it) {
      int kk = bkk + it * 16;
      __bf16 tmp[8];
      *(uint4*)tmp = *(const uint4*)(B + (size_t)(k0 + kk) * N + n0 + bnn);
#pragma unroll
      for (int i = 0; i < 8; ++i) sBt[st][(bnn + i) * 32 + kk] = tmp[i];
    }
  };

  stageA(0, 0);
  stageB(0, 0);
  async_wait0();
  __syncthreads();

  const int nk = K / 32;
  for (int kt = 0; kt < nk; ++kt) {
    const int st = kt & 1;

    // read current-stage fragments
    v16bf af[4], bq[4];
#pragma unroll
    for (int mi = 0; mi < 4; ++mi) {
      int row = wm + mi * 16 + lrow;
      af[mi] = make_frag(&sA[st][row * 32 + lhalf * 8],
                         &sA[st][row * 32 + lhalf * 8 + 16]);
    }
#pragma unroll
    for (int ni = 0; ni < 4; ++ni) {
      int col = wn + ni * 16 + lrow;
      bq[ni] = make_frag(&sBt[st][col * 32 + lhalf * 8],
                         &sBt[st][col * 32 + lhalf * 8 + 16]);
    }

    // prefetch next stage while computing
    if (kt + 1 < nk) {
      stageA((kt + 1) * 32, st ^ 1);
      stageB((kt + 1) * 32, st ^ 1);
    }

#pragma unroll
    for (int mi = 0; mi < 4; ++mi)
#pragma unroll
      for (int ni = 0; ni < 4; ++ni)
        acc[mi][ni] = __builtin_amdgcn_wmma_f32_16x16x32_bf16(
            false, af[mi], false, bq[ni], (short)0, acc[mi][ni], false, false);

    async_wait0();
    __syncthreads();
  }

#pragma unroll
  for (int mi = 0; mi < 4; ++mi)
#pragma unroll
    for (int ni = 0; ni < 4; ++ni)
#pragma unroll
      for (int i = 0; i < 8; ++i) {
        int row = m0 + wm + mi * 16 + i + 8 * lhalf;
        int col = n0 + wn + ni * 16 + lrow;
        C[(size_t)row * N + col] = acc[mi][ni][i];
      }
}

// ---------------------------------------------------------------------------
// Row-wise RMSNorm, fp32 in -> bf16 out
// ---------------------------------------------------------------------------
__global__ __launch_bounds__(256)
void rmsnorm_bf16(__bf16* __restrict__ dst, const float* __restrict__ src,
                  const float* __restrict__ w, int cols, int ld_src, int ld_dst) {
  const int row = blockIdx.x;
  const int tid = threadIdx.x;
  const float* x = src + (size_t)row * ld_src;
  float ss = 0.f;
  for (int c = tid; c < cols; c += 256) { float v = x[c]; ss += v * v; }
#pragma unroll
  for (int off = 16; off >= 1; off >>= 1) ss += __shfl_xor(ss, off, 32);
  __shared__ float red[8];
  if ((tid & 31) == 0) red[tid >> 5] = ss;
  __syncthreads();
  if (tid == 0) {
    float s2 = 0.f;
#pragma unroll
    for (int i = 0; i < 8; ++i) s2 += red[i];
    red[0] = rsqrtf(s2 / (float)cols + 1e-6f);
  }
  __syncthreads();
  float inv = red[0];
  __bf16* o = dst + (size_t)row * ld_dst;
  for (int c = tid; c < cols; c += 256) o[c] = (__bf16)(x[c] * inv * w[c]);
}

// ---------------------------------------------------------------------------
// RoPE (interleaved rotate-half, emb = concat(freqs, freqs))
// inv_freq[m] = 10000^(-m/32), m = j & 31
// ---------------------------------------------------------------------------
static __device__ __forceinline__ float rope_elem(const float* x64, int jj, float pos) {
  float xv  = x64[jj];
  float rot = (jj & 1) ? x64[jj - 1] : -x64[jj + 1];
  int   m   = jj & 31;
  float invf = __expf(-(float)m * (9.210340371976184f / 32.0f)); // ln(1e4)/32
  float ang = pos * invf;
  return xv * __cosf(ang) + rot * __sinf(ang);
}

#define ATT_SCALE 0.07216878364870323f   // 1/sqrt(192), folded into Q

// q fp32 [S][H*192] -> qfull bf16 [H][S][192] with RoPE on last 64 dims.
// Softmax scale is pre-applied here.
__global__ __launch_bounds__(256)
void build_q(__bf16* __restrict__ qfull, const float* __restrict__ qf) {
  const int s = blockIdx.x;
  for (int idx = threadIdx.x; idx < NHEAD * D_QK; idx += 256) {
    int h = idx / D_QK, d = idx % D_QK;
    const float* qrow = qf + (size_t)s * (NHEAD * D_QK) + h * D_QK;
    float out = (d < D_NOPE) ? qrow[d] : rope_elem(qrow + D_NOPE, d - D_NOPE, (float)s);
    qfull[((size_t)h * S_LEN + s) * D_QK + d] = (__bf16)(out * ATT_SCALE);
  }
}

// kvup fp32 [S][H*256], kv fp32 [S][KVPAD] -> kfull bf16 [H][S][192], v bf16 [H][S][128]
__global__ __launch_bounds__(256)
void build_kv(__bf16* __restrict__ kfull, __bf16* __restrict__ vbuf,
              const float* __restrict__ kvup, const float* __restrict__ kvf) {
  const int s = blockIdx.x;
  __shared__ float kro[D_ROPE];
  if (threadIdx.x < D_ROPE) {
    const float* kr = kvf + (size_t)s * KVPAD + KVLORA;
    kro[threadIdx.x] = rope_elem(kr, threadIdx.x, (float)s);
  }
  __syncthreads();
  for (int idx = threadIdx.x; idx < NHEAD * 256; idx += 256) {
    int h = idx >> 8, d = idx & 255;
    float v = kvup[(size_t)s * (NHEAD * 256) + h * 256 + d];
    if (d < D_NOPE)
      kfull[((size_t)h * S_LEN + s) * D_QK + d] = (__bf16)v;
    else
      vbuf[((size_t)h * S_LEN + s) * D_V + (d - D_NOPE)] = (__bf16)v;
  }
  for (int idx = threadIdx.x; idx < NHEAD * D_ROPE; idx += 256) {
    int h = idx / D_ROPE, jj = idx % D_ROPE;
    kfull[((size_t)h * S_LEN + s) * D_QK + D_NOPE + jj] = (__bf16)kro[jj];
  }
}

// ---------------------------------------------------------------------------
// Flash attention, causal. 4 waves, Br=64 (16 q rows/wave), Bc=64.
// Double-buffered K/V staging (async K copy). O bf16 [S][H*128].
// ---------------------------------------------------------------------------
__global__ __launch_bounds__(128)
void flash_attn(__bf16* __restrict__ O, const __bf16* __restrict__ Qf,
                const __bf16* __restrict__ Kf, const __bf16* __restrict__ Vf) {
  __shared__ __align__(16) __bf16 sQ[64 * D_QK];
  __shared__ __align__(16) __bf16 sK[2][64 * D_QK];
  __shared__ __align__(16) __bf16 sVt[2][D_V * 64];     // [d][key]
  __shared__ __align__(16) __bf16 sP[4][16 * 64];       // per-wave P scratch

  const int h    = blockIdx.y;
  const int qt   = blockIdx.x;
  const int qb   = qt * 64;
  const int tid  = threadIdx.x;
  const int wave = tid >> 5;
  const int lane = tid & 31;
  const int lrow = lane & 15;
  const int lhalf = lane >> 4;
  const int r0 = wave * 16;

  const __bf16* Qh = Qf + (size_t)h * S_LEN * D_QK;
  const __bf16* Kh = Kf + (size_t)h * S_LEN * D_QK;
  const __bf16* Vh = Vf + (size_t)h * S_LEN * D_V;

  auto stageKV = [&](int j, int st) {
    // K tile: async row-major copy
    for (int c = tid; c < 64 * D_QK / 8; c += 128) {
      int row = c / 24, cg = (c % 24) * 8;
      async_b128(Kh + (size_t)(j * 64 + row) * D_QK + cg, &sK[st][row * D_QK + cg]);
    }
    // V tile: transpose into LDS
    for (int c = tid; c < 64 * D_V / 8; c += 128) {
      int row = c >> 4, dg = (c & 15) * 8;
      __bf16 tmp[8];
      *(uint4*)tmp = *(const uint4*)(Vh + (size_t)(j * 64 + row) * D_V + dg);
#pragma unroll
      for (int i = 0; i < 8; ++i) sVt[st][(dg + i) * 64 + row] = tmp[i];
    }
  };

  // stage Q tile (async) + first K/V tile
  for (int c = tid; c < 64 * D_QK / 8; c += 128) {
    int row = c / 24, cg = (c % 24) * 8;
    async_b128(Qh + (size_t)(qb + row) * D_QK + cg, &sQ[row * D_QK + cg]);
  }
  stageKV(0, 0);
  async_wait0();
  __syncthreads();

  // cache Q fragments (A-operand layout), K = 192 = 6 x 32
  v16bf qfrag[6];
#pragma unroll
  for (int ks = 0; ks < 6; ++ks) {
    int row = r0 + lrow;
    int base = ks * 32 + lhalf * 8;
    qfrag[ks] = make_frag(&sQ[row * D_QK + base], &sQ[row * D_QK + base + 16]);
  }

  float mrow[8], lsum[8];
  v8f   oacc[8];
#pragma unroll
  for (int i = 0; i < 8; ++i) { mrow[i] = -3.0e38f; lsum[i] = 0.f; oacc[i] = (v8f)0.0f; }

  for (int j = 0; j <= qt; ++j) {
    const int st = j & 1;

    // prefetch next K/V tile while computing this one
    if (j < qt) stageKV(j + 1, st ^ 1);

    // scores S = Q K^T  (B-operand = K rows directly)
    v8f sacc[4];
#pragma unroll
    for (int t = 0; t < 4; ++t) sacc[t] = (v8f)0.0f;
#pragma unroll
    for (int t = 0; t < 4; ++t)
#pragma unroll
      for (int ks = 0; ks < 6; ++ks) {
        int key = t * 16 + lrow;
        int base = ks * 32 + lhalf * 8;
        v16bf bfr = make_frag(&sK[st][key * D_QK + base],
                              &sK[st][key * D_QK + base + 16]);
        sacc[t] = __builtin_amdgcn_wmma_f32_16x16x32_bf16(
            false, qfrag[ks], false, bfr, (short)0, sacc[t], false, false);
      }

    // causal mask (diagonal tile only) + row max
    float newm[8];
#pragma unroll
    for (int i = 0; i < 8; ++i) newm[i] = mrow[i];
#pragma unroll
    for (int t = 0; t < 4; ++t)
#pragma unroll
      for (int i = 0; i < 8; ++i) {
        float v = sacc[t][i];
        if (j == qt) {
          int row_g = qb + r0 + i + 8 * lhalf;
          int key_g = j * 64 + t * 16 + lrow;
          if (key_g > row_g) v = -3.0e38f;
        }
        sacc[t][i] = v;
        newm[i] = fmaxf(newm[i], v);
      }
#pragma unroll
    for (int off = 1; off <= 8; off <<= 1)
#pragma unroll
      for (int i = 0; i < 8; ++i) newm[i] = fmaxf(newm[i], __shfl_xor(newm[i], off, 32));

    float alpha[8], rsum[8];
#pragma unroll
    for (int i = 0; i < 8; ++i) { alpha[i] = __expf(mrow[i] - newm[i]); rsum[i] = 0.f; }

    // p = exp(s - m), write to per-wave LDS scratch (A-operand source)
#pragma unroll
    for (int t = 0; t < 4; ++t)
#pragma unroll
      for (int i = 0; i < 8; ++i) {
        float p = __expf(sacc[t][i] - newm[i]);
        rsum[i] += p;
        sP[wave][(i + 8 * lhalf) * 64 + t * 16 + lrow] = (__bf16)p;
      }
#pragma unroll
    for (int off = 1; off <= 8; off <<= 1)
#pragma unroll
      for (int i = 0; i < 8; ++i) rsum[i] += __shfl_xor(rsum[i], off, 32);

#pragma unroll
    for (int i = 0; i < 8; ++i) {
      lsum[i] = lsum[i] * alpha[i] + rsum[i];
      mrow[i] = newm[i];
    }
#pragma unroll
    for (int td = 0; td < 8; ++td)
#pragma unroll
      for (int i = 0; i < 8; ++i) oacc[td][i] *= alpha[i];

    // same-wave LDS write->read ordering for sP
    asm volatile("s_wait_dscnt 0" ::: "memory");

    // O += P @ V   (B-operand = V^T in LDS)
#pragma unroll
    for (int ks2 = 0; ks2 < 2; ++ks2) {
      int pbase = ks2 * 32 + lhalf * 8;
      v16bf pa = make_frag(&sP[wave][lrow * 64 + pbase],
                           &sP[wave][lrow * 64 + pbase + 16]);
#pragma unroll
      for (int td = 0; td < 8; ++td) {
        int d = td * 16 + lrow;
        int vbase = ks2 * 32 + lhalf * 8;
        v16bf vb = make_frag(&sVt[st][d * 64 + vbase],
                             &sVt[st][d * 64 + vbase + 16]);
        oacc[td] = __builtin_amdgcn_wmma_f32_16x16x32_bf16(
            false, pa, false, vb, (short)0, oacc[td], false, false);
      }
    }

    // next stage's async K + ds-store V must land before the barrier
    async_wait0();
    __syncthreads();
  }

  // epilogue: normalize + store bf16 [s][h*128 + d]
#pragma unroll
  for (int td = 0; td < 8; ++td)
#pragma unroll
    for (int i = 0; i < 8; ++i) {
      int s = qb + r0 + i + 8 * lhalf;
      int d = td * 16 + lrow;
      O[(size_t)s * (NHEAD * D_V) + h * D_V + d] = (__bf16)(oacc[td][i] / lsum[i]);
    }
}

// ---------------------------------------------------------------------------
// Host launch
// ---------------------------------------------------------------------------
extern "C" void kernel_launch(void* const* d_in, const int* in_sizes, int n_in,
                              void* d_out, int out_size, void* d_ws, size_t ws_size,
                              hipStream_t stream) {
  const float* x     = (const float*)d_in[0];
  const float* w_qa  = (const float*)d_in[1];
  const float* qnw   = (const float*)d_in[2];
  const float* w_qb  = (const float*)d_in[3];
  const float* w_kva = (const float*)d_in[4];
  const float* kvnw  = (const float*)d_in[5];
  const float* w_kvb = (const float*)d_in[6];
  const float* w_o   = (const float*)d_in[7];

  char* ws = (char*)d_ws;
  size_t off = 0;
  auto take = [&](size_t bytes) -> char* {
    char* p = ws + off;
    off = (off + bytes + 255) & ~(size_t)255;
    return p;
  };

  __bf16* xb    = (__bf16*)take((size_t)S_LEN * HID * 2);
  __bf16* wqab  = (__bf16*)take((size_t)HID * QLORA * 2);
  __bf16* wqbb  = (__bf16*)take((size_t)QLORA * NHEAD * D_QK * 2);
  __bf16* wkvab = (__bf16*)take((size_t)HID * KVPAD * 2);       // padded to 640
  __bf16* wkvbb = (__bf16*)take((size_t)KVLORA * NHEAD * 256 * 2);
  __bf16* wob   = (__bf16*)take((size_t)(NHEAD * D_V) * HID * 2);
  float*  qa    = (float*)take((size_t)S_LEN * QLORA * 4);
  float*  kvf   = (float*)take((size_t)S_LEN * KVPAD * 4);      // padded to 640
  __bf16* qan   = (__bf16*)take((size_t)S_LEN * QLORA * 2);
  __bf16* ckvn  = (__bf16*)take((size_t)S_LEN * KVLORA * 2);
  float*  qf    = (float*)take((size_t)S_LEN * NHEAD * D_QK * 4);
  float*  kvup  = (float*)take((size_t)S_LEN * NHEAD * 256 * 4);
  __bf16* qfull = (__bf16*)take((size_t)NHEAD * S_LEN * D_QK * 2);
  __bf16* kfull = (__bf16*)take((size_t)NHEAD * S_LEN * D_QK * 2);
  __bf16* vbuf  = (__bf16*)take((size_t)NHEAD * S_LEN * D_V * 2);
  __bf16* attn  = (__bf16*)take((size_t)S_LEN * NHEAD * D_V * 2);

  auto cvt = [&](__bf16* d, const float* s, int n) {
    cvt_f32_bf16<<<dim3((n / 4 + 255) / 256), dim3(256), 0, stream>>>(d, s, n);
  };
  cvt(xb,    x,     S_LEN * HID);
  cvt(wqab,  w_qa,  HID * QLORA);
  cvt(wqbb,  w_qb,  QLORA * NHEAD * D_QK);
  cvt(wkvbb, w_kvb, KVLORA * NHEAD * 256);
  cvt(wob,   w_o,   NHEAD * D_V * HID);
  {
    int total = HID * KVPAD;
    cvt_pad_bf16<<<dim3((total + 255) / 256), dim3(256), 0, stream>>>(
        wkvab, w_kva, HID, KVLORA + D_ROPE, KVPAD);
  }

  auto gemm = [&](float* C, const __bf16* A, const __bf16* B, int M, int N, int K) {
    gemm_bf16<<<dim3(N / 128, M / 256), dim3(256), 0, stream>>>(C, A, B, M, N, K);
  };

  gemm(qa,  xb, wqab,  S_LEN, QLORA, HID);   // x @ w_qa
  gemm(kvf, xb, wkvab, S_LEN, KVPAD, HID);   // x @ w_kva (padded N)

  rmsnorm_bf16<<<dim3(S_LEN), dim3(256), 0, stream>>>(qan, qa, qnw, QLORA, QLORA, QLORA);
  rmsnorm_bf16<<<dim3(S_LEN), dim3(256), 0, stream>>>(ckvn, kvf, kvnw, KVLORA,
                                                      KVPAD, KVLORA);

  gemm(qf,   qan,  wqbb,  S_LEN, NHEAD * D_QK, QLORA);  // q up-proj
  gemm(kvup, ckvn, wkvbb, S_LEN, NHEAD * 256,  KVLORA); // kv up-proj

  build_q<<<dim3(S_LEN), dim3(256), 0, stream>>>(qfull, qf);
  build_kv<<<dim3(S_LEN), dim3(256), 0, stream>>>(kfull, vbuf, kvup, kvf);

  flash_attn<<<dim3(S_LEN / 64, NHEAD), dim3(128), 0, stream>>>(attn, qfull, kfull, vbuf);

  gemm((float*)d_out, attn, wob, S_LEN, HID, NHEAD * D_V); // output projection
}
